// BlockMoba_27195732918647
// MI455X (gfx1250) — matrix operand order repaired and checked by reading, not compiled, not run on hardware
//
#include <hip/hip_runtime.h>

// ---------------------------------------------------------------------------
// MI455X (gfx1250, wave32) fused transformer block:
//   rmsnorm -> projection-free flash attention (bf16 WMMA) -> residual
//   rmsnorm -> top2-of-8 gate -> SwiGLU experts + shared SiLU MLP (bf16 WMMA)
// GEMMs: v_wmma_f32_16x16x32_bf16, K-step 64, packed bf16 LDS staging
// (v_cvt_pk_bf16_f32), async global->LDS b128 (ASYNCcnt) for expert h-tiles,
// Tensor Data Mover (TENSORcnt) for the shared-expert h-tiles.
// ---------------------------------------------------------------------------

typedef __bf16 bf16;
typedef __attribute__((ext_vector_type(2)))  __bf16 v2bf;
typedef __attribute__((ext_vector_type(16))) __bf16 v16bf;
typedef __attribute__((ext_vector_type(8)))  float  v8f;
typedef unsigned int u32x4 __attribute__((ext_vector_type(4)));
typedef int          i32x4 __attribute__((ext_vector_type(4)));
typedef int          i32x8 __attribute__((ext_vector_type(8)));

#define SEQ   2048
#define DIM   768
#define NHEAD 12
#define HDIM  64
#define NEXP  8
#define IDIM  1024
#define ISDIM 2048
#define LDT   72   // LDS tile row stride in bf16 elements (144B, 16B aligned)

__device__ __forceinline__ v8f wmma_bf16(v16bf a, v16bf b, v8f c) {
  return __builtin_amdgcn_wmma_f32_16x16x32_bf16(false, a, false, b, (short)0, c,
                                                 false, false);
}

// single v_cvt_pk_bf16_f32
__device__ __forceinline__ unsigned pack_bf16_2(float lo, float hi) {
  v2bf v;
  v[0] = (bf16)lo;
  v[1] = (bf16)hi;
  return __builtin_bit_cast(unsigned, v);
}

// A-matrix 16x32 bf16 fragment: lane m = lane&15,
// VGPR pair v -> K = 2v + 16*(lane>>4)
__device__ __forceinline__ v16bf frag_a(const bf16* p, int ld, int lane) {
  int m = lane & 15, grp = lane >> 4;
  v16bf f;
#pragma unroll
  for (int v = 0; v < 8; ++v) {
    int k0 = 2 * v + 16 * grp;
    f[2 * v]     = p[m * ld + k0];
    f[2 * v + 1] = p[m * ld + k0 + 1];
  }
  return f;
}

// B-matrix 32x16 bf16 fragment from [N][K] (row-major weight) storage:
// N = lane&15, element e -> K = e + 16*(lane>>4)
__device__ __forceinline__ v16bf frag_b_nk(const bf16* p, int ld, int lane) {
  int n = lane & 15, grp = lane >> 4;
  v16bf f;
#pragma unroll
  for (int e = 0; e < 16; ++e) f[e] = p[n * ld + e + 16 * grp];
  return f;
}

// B-matrix fragment from [K][N] storage (used for P*V where V rows are keys)
__device__ __forceinline__ v16bf frag_b_kn(const bf16* p, int ld, int lane) {
  int n = lane & 15, grp = lane >> 4;
  v16bf f;
#pragma unroll
  for (int e = 0; e < 16; ++e) f[e] = p[(e + 16 * grp) * ld + n];
  return f;
}

__device__ __forceinline__ float red_sum32(float v) {
#pragma unroll
  for (int off = 16; off >= 1; off >>= 1) v += __shfl_xor(v, off, 32);
  return v;
}
__device__ __forceinline__ float red_sum16(float v) {
#pragma unroll
  for (int off = 8; off >= 1; off >>= 1) v += __shfl_xor(v, off, 16);
  return v;
}
__device__ __forceinline__ float red_max16(float v) {
#pragma unroll
  for (int off = 8; off >= 1; off >>= 1) v = fmaxf(v, __shfl_xor(v, off, 16));
  return v;
}

// ---------------------------------------------------------------------------
// RMSNorm: one wave per token
// ---------------------------------------------------------------------------
__global__ void k_rmsnorm(const float* __restrict__ x, const float* __restrict__ w,
                          float* __restrict__ o) {
  int wave = threadIdx.x >> 5, lane = threadIdx.x & 31;
  int n = blockIdx.x * 8 + wave;
  const float* xr = x + (size_t)n * DIM;
  float ss = 0.f;
  for (int d = lane; d < DIM; d += 32) { float v = xr[d]; ss += v * v; }
  ss = red_sum32(ss);
  float r = rsqrtf(ss * (1.0f / DIM) + 1e-5f);
  float* orow = o + (size_t)n * DIM;
  for (int d = lane; d < DIM; d += 32) orow[d] = xr[d] * r * w[d];
}

// ---------------------------------------------------------------------------
// Flash attention, Q=K=V=xn. 1 wave per (16-row q tile, head).
// k-tile = 32 keys -> P*V is exactly one K=32 WMMA depth.
// ---------------------------------------------------------------------------
__global__ void k_attn(const float* __restrict__ x, const float* __restrict__ xn,
                       float* __restrict__ outb) {
  int qt = blockIdx.x, head = blockIdx.y;
  int lane = threadIdx.x & 31;
  int grp = lane >> 4, nl = lane & 15;

  __shared__ bf16 sQ[16][LDT];
  __shared__ bf16 sKV[32][LDT];
  __shared__ bf16 sP[16][40];

  // stage Q tile (16 x 64) as packed bf16 pairs
#pragma unroll
  for (int t = 0; t < 16; ++t) {
    int i = lane + 32 * t;           // u32 index over 16x32 packed tile
    int r = i >> 5, cu = i & 31;
    float2 v = *(const float2*)&xn[(size_t)(qt * 16 + r) * DIM + head * HDIM + cu * 2];
    *(unsigned*)&sQ[r][cu * 2] = pack_bf16_2(v.x, v.y);
  }
  __syncthreads();
  v16bf qa0 = frag_a(&sQ[0][0], LDT, lane);
  v16bf qa1 = frag_a(&sQ[0][32], LDT, lane);

  v8f o[4];
  float mi[8], li[8];
#pragma unroll
  for (int t = 0; t < 4; ++t)
#pragma unroll
    for (int r = 0; r < 8; ++r) o[t][r] = 0.f;
#pragma unroll
  for (int r = 0; r < 8; ++r) { mi[r] = -3e38f; li[r] = 0.f; }

  for (int kt = 0; kt < SEQ / 32; ++kt) {
    __syncthreads();
#pragma unroll
    for (int t = 0; t < 32; ++t) {   // 32x32 packed u32 tile
      int i = lane + 32 * t;
      int r = i >> 5, cu = i & 31;
      float2 v = *(const float2*)&xn[(size_t)(kt * 32 + r) * DIM + head * HDIM + cu * 2];
      *(unsigned*)&sKV[r][cu * 2] = pack_bf16_2(v.x, v.y);
    }
    __syncthreads();

    // S = (Q K^T) / sqrt(64): two 16x16 N-tiles covering 32 keys
    v8f s[2];
#pragma unroll
    for (int j = 0; j < 2; ++j) {
      v16bf kb0 = frag_b_nk(&sKV[j * 16][0], LDT, lane);
      v16bf kb1 = frag_b_nk(&sKV[j * 16][32], LDT, lane);
      v8f z = {0.f, 0.f, 0.f, 0.f, 0.f, 0.f, 0.f, 0.f};
      z = wmma_bf16(qa0, kb0, z);
      z = wmma_bf16(qa1, kb1, z);
#pragma unroll
      for (int r = 0; r < 8; ++r) z[r] *= 0.125f;
      s[j] = z;
    }

    // online softmax over the 32-key slab (row = r + 8*grp)
    float p0[8], p1[8];
#pragma unroll
    for (int r = 0; r < 8; ++r) {
      float mx = red_max16(fmaxf(s[0][r], s[1][r]));
      float mn = fmaxf(mi[r], mx);
      float sc = __expf(mi[r] - mn);
      p0[r] = __expf(s[0][r] - mn);
      p1[r] = __expf(s[1][r] - mn);
      float rs = red_sum16(p0[r] + p1[r]);
      li[r] = li[r] * sc + rs;
      mi[r] = mn;
#pragma unroll
      for (int t = 0; t < 4; ++t) o[t][r] *= sc;
    }

    // round-trip P through LDS: C-layout -> A-fragment layout
#pragma unroll
    for (int r = 0; r < 8; ++r) {
      sP[r + 8 * grp][nl]      = (bf16)p0[r];
      sP[r + 8 * grp][nl + 16] = (bf16)p1[r];
    }
    __syncthreads();
    v16bf pa = frag_a(&sP[0][0], 40, lane);
#pragma unroll
    for (int t = 0; t < 4; ++t) {
      v16bf vb = frag_b_kn(&sKV[0][t * 16], LDT, lane);
      o[t] = wmma_bf16(pa, vb, o[t]);
    }
  }

  // epilogue: out = x + attn
#pragma unroll
  for (int r = 0; r < 8; ++r) {
    float inv = 1.0f / li[r];
    int q = qt * 16 + r + 8 * grp;
#pragma unroll
    for (int t = 0; t < 4; ++t) {
      int c = head * HDIM + t * 16 + nl;
      outb[(size_t)q * DIM + c] = x[(size_t)q * DIM + c] + o[t][r] * inv;
    }
  }
}

// ---------------------------------------------------------------------------
// Gate: softmax over 8 experts, top-2 -> dense combine weights. Wave/token.
// ---------------------------------------------------------------------------
__global__ void k_gate(const float* __restrict__ xf, const float* __restrict__ gw,
                       float* __restrict__ cmb) {
  int wave = threadIdx.x >> 5, lane = threadIdx.x & 31;
  int n = blockIdx.x * 8 + wave;
  const float* xr = xf + (size_t)n * DIM;
  float lg[NEXP];
#pragma unroll
  for (int e = 0; e < NEXP; ++e) {
    float p = 0.f;
    for (int d = lane; d < DIM; d += 32) p += xr[d] * gw[e * DIM + d];
    lg[e] = red_sum32(p);
  }
  float mx = lg[0];
#pragma unroll
  for (int e = 1; e < NEXP; ++e) mx = fmaxf(mx, lg[e]);
  float pr[NEXP], ssum = 0.f;
#pragma unroll
  for (int e = 0; e < NEXP; ++e) { pr[e] = __expf(lg[e] - mx); ssum += pr[e]; }
  float inv = 1.0f / ssum;
#pragma unroll
  for (int e = 0; e < NEXP; ++e) pr[e] *= inv;
  int i1 = 0;
#pragma unroll
  for (int e = 1; e < NEXP; ++e) if (pr[e] > pr[i1]) i1 = e;
  int i2 = (i1 == 0) ? 1 : 0;
#pragma unroll
  for (int e = 0; e < NEXP; ++e) if (e != i1 && pr[e] > pr[i2]) i2 = e;
  if (lane == 0) {
#pragma unroll
    for (int e = 0; e < NEXP; ++e)
      cmb[n * NEXP + e] = (e == i1 || e == i2) ? fmaxf(pr[e], 1e-7f) : 0.f;
  }
}

// ---------------------------------------------------------------------------
// h = silu(xf*W1^T + b1) [* (xf*W3^T + b3) if gated], bf16 output.
// 256 threads; block computes 64(M) x 64(N); K-step 64 (4/8 WMMA per stage).
// Skips whole tile when all 64 tokens have zero combine weight (top-2 mask).
// ---------------------------------------------------------------------------
__global__ void k_moe_h(const float* __restrict__ xf,
                        const float* __restrict__ w1, const float* __restrict__ b1,
                        const float* __restrict__ w3, const float* __restrict__ b3,
                        const float* __restrict__ cmb, int expert, int kdim,
                        bf16* __restrict__ hout, int ldh, int gated) {
  __shared__ bf16 sA[64][LDT];
  __shared__ bf16 sB1[64][LDT];
  __shared__ bf16 sB3[64][LDT];
  __shared__ int s_any;

  int tid = threadIdx.x;
  int n0 = blockIdx.x * 64, m0 = blockIdx.y * 64;

  if (cmb) {
    if (tid == 0) s_any = 0;
    __syncthreads();
    if (tid < 64 && cmb[(m0 + tid) * NEXP + expert] != 0.f) s_any = 1;
    __syncthreads();
    if (!s_any) return;
  }

  int wave = tid >> 5, lane = tid & 31;
  int wm = wave & 3, wn = wave >> 2;
  v8f a0 = {0.f,0.f,0.f,0.f,0.f,0.f,0.f,0.f}, a1 = a0, g0 = a0, g1 = a0;

  int nk = kdim / 64;
  for (int kt = 0; kt < nk; ++kt) {
    // 64x64 tile = 2048 packed u32 -> 8 per thread, per matrix
#pragma unroll
    for (int i = 0; i < 8; ++i) {
      int id = tid + 256 * i;
      int r = id >> 5, cu = id & 31;
      float2 va = *(const float2*)&xf[(size_t)(m0 + r) * kdim + kt * 64 + cu * 2];
      *(unsigned*)&sA[r][cu * 2] = pack_bf16_2(va.x, va.y);
      float2 vb = *(const float2*)&w1[(size_t)(n0 + r) * kdim + kt * 64 + cu * 2];
      *(unsigned*)&sB1[r][cu * 2] = pack_bf16_2(vb.x, vb.y);
      if (gated) {
        float2 vc = *(const float2*)&w3[(size_t)(n0 + r) * kdim + kt * 64 + cu * 2];
        *(unsigned*)&sB3[r][cu * 2] = pack_bf16_2(vc.x, vc.y);
      }
    }
    if (kt + 1 < nk) {
      __builtin_prefetch(&xf[(size_t)(m0 + (tid & 63)) * kdim + (kt + 1) * 64], 0, 1);
      __builtin_prefetch(&w1[(size_t)(n0 + (tid & 63)) * kdim + (kt + 1) * 64], 0, 1);
    }
    __syncthreads();
    v16bf af0 = frag_a(&sA[wm * 16][0], LDT, lane);
    v16bf af1 = frag_a(&sA[wm * 16][32], LDT, lane);
#pragma unroll
    for (int kk = 0; kk < 2; ++kk) {
      v16bf af = kk ? af1 : af0;
      a0 = wmma_bf16(af, frag_b_nk(&sB1[wn * 32][kk * 32], LDT, lane), a0);
      a1 = wmma_bf16(af, frag_b_nk(&sB1[wn * 32 + 16][kk * 32], LDT, lane), a1);
      if (gated) {
        g0 = wmma_bf16(af, frag_b_nk(&sB3[wn * 32][kk * 32], LDT, lane), g0);
        g1 = wmma_bf16(af, frag_b_nk(&sB3[wn * 32 + 16][kk * 32], LDT, lane), g1);
      }
    }
    __syncthreads();
  }

  int grp = lane >> 4, nl = lane & 15;
#pragma unroll
  for (int sub = 0; sub < 2; ++sub) {
    v8f av = sub ? a1 : a0;
    v8f gv = sub ? g1 : g0;
#pragma unroll
    for (int r = 0; r < 8; ++r) {
      int m = m0 + wm * 16 + r + 8 * grp;
      int n = n0 + wn * 32 + sub * 16 + nl;
      float a = av[r] + b1[n];
      float h = a / (1.0f + __expf(-a));
      if (gated) h *= (gv[r] + b3[n]);
      hout[(size_t)m * ldh + n] = (bf16)h;
    }
  }
}

// ---------------------------------------------------------------------------
// y += cmb[:,e] * (h * W2^T + b2)   (cmb == nullptr -> scale 1, shared fc2)
// A tile (bf16 h): per-lane async global->LDS b128 (ASYNCcnt), or one
// Tensor Data Mover 2-D tile descriptor with LDS row padding (TENSORcnt).
// ---------------------------------------------------------------------------
__global__ void k_moe_out(const bf16* __restrict__ h, int ldh,
                          const float* __restrict__ w2, const float* __restrict__ b2,
                          const float* __restrict__ cmb, int expert, int kdim,
                          float* __restrict__ y, int use_tdm) {
  __shared__ bf16 sA[64][LDT];
  __shared__ bf16 sB[64][LDT];
  __shared__ int s_any;

  int tid = threadIdx.x;
  int n0 = blockIdx.x * 64, m0 = blockIdx.y * 64;

  if (cmb) {
    if (tid == 0) s_any = 0;
    __syncthreads();
    if (tid < 64 && cmb[(m0 + tid) * NEXP + expert] != 0.f) s_any = 1;
    __syncthreads();
    if (!s_any) return;
  }

  int wave = tid >> 5, lane = tid & 31;
  int wm = wave & 3, wn = wave >> 2;
  v8f a0 = {0.f,0.f,0.f,0.f,0.f,0.f,0.f,0.f}, a1 = a0;

  int nk = kdim / 64;
  for (int kt = 0; kt < nk; ++kt) {
    if (use_tdm) {
      if (wave == 0) {
        // TDM D# (ISA 8.3/8.4): 64x64 bf16 tile, row stride ldh elements,
        // LDS pad 4 DWORDs after every 32 DWORDs -> 72-element LDS rows.
        unsigned lds = (unsigned)(uintptr_t)&sA[0][0];
        unsigned long long ga =
            (unsigned long long)(uintptr_t)(h + (size_t)m0 * ldh + kt * 64);
        u32x4 g0;
        g0[0] = 1u;                                   // count=1, load descriptor
        g0[1] = lds;                                  // lds_addr
        g0[2] = (unsigned)(ga & 0xFFFFFFFFull);       // global_addr[31:0]
        g0[3] = (unsigned)((ga >> 32) & 0x1FFFFFFull) // global_addr[56:32]
                | (2u << 30);                         // type=2 (image)
        i32x8 g1;
        g1[0] = (int)(0x10000u                        // data_size=1 (2B)
                | (1u << 20)                          // pad_enable
                | (4u << 22)                          // pad_interval: 32 DWORDs
                | (3u << 25));                        // pad_amount: 4 DWORDs
        g1[1] = (int)((ldh & 0xFFFF) << 16);          // tensor_dim0[15:0]
        g1[2] = (int)((ldh >> 16) | ((SEQ & 0xFFFF) << 16)); // dim0 hi | dim1 lo
        g1[3] = (int)((SEQ >> 16) | (64 << 16));      // dim1 hi | tile_dim0=64
        g1[4] = 64;                                   // tile_dim1=64, tile_dim2=0
        g1[5] = (int)ldh;                             // tensor_dim0_stride lo
        g1[6] = 0;
        g1[7] = 0;
        i32x4 gz4 = {0, 0, 0, 0};
        i32x8 gz8 = {0, 0, 0, 0, 0, 0, 0, 0};
        __builtin_amdgcn_tensor_load_to_lds(g0, g1, gz4, gz4, gz8, 0);
      }
    } else {
      // A tile: 64 rows x 128B = 512 x 16B chunks; 2 per thread, async to LDS
#pragma unroll
      for (int i = 0; i < 2; ++i) {
        int id = tid + 256 * i;
        int r = id >> 3, ch = id & 7;
        unsigned lds = (unsigned)(uintptr_t)&sA[r][ch * 8];
        const bf16* g = h + (size_t)(m0 + r) * ldh + kt * 64 + ch * 8;
        asm volatile("global_load_async_to_lds_b128 %0, %1, off"
                     :: "v"(lds), "v"(g) : "memory");
      }
    }
    // B tile: f32 weights -> packed bf16
#pragma unroll
    for (int i = 0; i < 8; ++i) {
      int id = tid + 256 * i;
      int r = id >> 5, cu = id & 31;
      float2 vb = *(const float2*)&w2[(size_t)(n0 + r) * kdim + kt * 64 + cu * 2];
      *(unsigned*)&sB[r][cu * 2] = pack_bf16_2(vb.x, vb.y);
    }
    if (kt + 1 < nk)
      __builtin_prefetch(&w2[(size_t)(n0 + (tid & 63)) * kdim + (kt + 1) * 64], 0, 1);
    if (use_tdm) {
      if (wave == 0) __builtin_amdgcn_s_wait_tensorcnt(0);
    } else {
      asm volatile("s_wait_asynccnt 0x0" ::: "memory");
    }
    __syncthreads();
    v16bf af0 = frag_a(&sA[wm * 16][0], LDT, lane);
    v16bf af1 = frag_a(&sA[wm * 16][32], LDT, lane);
#pragma unroll
    for (int kk = 0; kk < 2; ++kk) {
      v16bf af = kk ? af1 : af0;
      a0 = wmma_bf16(af, frag_b_nk(&sB[wn * 32][kk * 32], LDT, lane), a0);
      a1 = wmma_bf16(af, frag_b_nk(&sB[wn * 32 + 16][kk * 32], LDT, lane), a1);
    }
    __syncthreads();
  }

  int grp = lane >> 4, nl = lane & 15;
#pragma unroll
  for (int sub = 0; sub < 2; ++sub) {
    v8f av = sub ? a1 : a0;
#pragma unroll
    for (int r = 0; r < 8; ++r) {
      int m = m0 + wm * 16 + r + 8 * grp;
      int n = n0 + wn * 32 + sub * 16 + nl;
      float scale = cmb ? cmb[m * NEXP + expert] : 1.0f;
      y[(size_t)m * DIM + n] += (av[r] + b2[n]) * scale;
    }
  }
}

__global__ void k_zero(float* __restrict__ p, int n) {
  int i = blockIdx.x * blockDim.x + threadIdx.x;
  if (i < n) p[i] = 0.f;
}

__global__ void k_final(const float* __restrict__ a, const float* __restrict__ b,
                        float* __restrict__ o, int n) {
  int i = blockIdx.x * blockDim.x + threadIdx.x;
  if (i < n) o[i] = a[i] + b[i];
}

// ---------------------------------------------------------------------------
extern "C" void kernel_launch(void* const* d_in, const int* in_sizes, int n_in,
                              void* d_out, int out_size, void* d_ws, size_t ws_size,
                              hipStream_t stream) {
  const float* x       = (const float*)d_in[0];
  const float* norm1_w = (const float*)d_in[1];
  const float* norm3_w = (const float*)d_in[2];
  const float* gate_w  = (const float*)d_in[3];
  const float* w1      = (const float*)d_in[4];
  const float* b1      = (const float*)d_in[5];
  const float* w2      = (const float*)d_in[6];
  const float* b2      = (const float*)d_in[7];
  const float* w3      = (const float*)d_in[8];
  const float* b3      = (const float*)d_in[9];
  const float* fc1_w   = (const float*)d_in[10];
  const float* fc1_b   = (const float*)d_in[11];
  const float* fc2_w   = (const float*)d_in[12];
  const float* fc2_b   = (const float*)d_in[13];
  float* out = (float*)d_out;

  char* p = (char*)d_ws;
  float* xn  = (float*)p;  p += (size_t)SEQ * DIM * sizeof(float);
  float* ob  = (float*)p;  p += (size_t)SEQ * DIM * sizeof(float);
  float* xf  = (float*)p;  p += (size_t)SEQ * DIM * sizeof(float);
  float* y   = (float*)p;  p += (size_t)SEQ * DIM * sizeof(float);
  float* cmb = (float*)p;  p += (size_t)SEQ * NEXP * sizeof(float);
  bf16*  hb  = (bf16*)p;   // SEQ x ISDIM bf16

  const int NTOT = SEQ * DIM;

  k_rmsnorm<<<SEQ / 8, 256, 0, stream>>>(x, norm1_w, xn);
  k_attn<<<dim3(SEQ / 16, NHEAD), 32, 0, stream>>>(x, xn, ob);
  k_rmsnorm<<<SEQ / 8, 256, 0, stream>>>(ob, norm3_w, xf);
  k_gate<<<SEQ / 8, 256, 0, stream>>>(xf, gate_w, cmb);
  k_zero<<<(NTOT + 255) / 256, 256, 0, stream>>>(y, NTOT);

  for (int e = 0; e < NEXP; ++e) {
    k_moe_h<<<dim3(IDIM / 64, SEQ / 64), 256, 0, stream>>>(
        xf, w1 + (size_t)e * IDIM * DIM, b1 + e * IDIM,
        w3 + (size_t)e * IDIM * DIM, b3 + e * IDIM,
        cmb, e, DIM, hb, IDIM, 1);
    k_moe_out<<<dim3(DIM / 64, SEQ / 64), 256, 0, stream>>>(
        hb, IDIM, w2 + (size_t)e * DIM * IDIM, b2 + e * DIM,
        cmb, e, IDIM, y, /*use_tdm=*/0);
  }

  // shared expert: z = silu(xf*fc1^T + fc1_b) * fc2^T + fc2_b, added into y
  k_moe_h<<<dim3(ISDIM / 64, SEQ / 64), 256, 0, stream>>>(
      xf, fc1_w, fc1_b, nullptr, nullptr, nullptr, 0, DIM, hb, ISDIM, 0);
  k_moe_out<<<dim3(DIM / 64, SEQ / 64), 256, 0, stream>>>(
      hb, ISDIM, fc2_w, fc2_b, nullptr, 0, ISDIM, y, /*use_tdm=*/1);

  k_final<<<(NTOT + 255) / 256, 256, 0, stream>>>(ob, y, out, NTOT);
}
